// AdvancedGNNOptimizer_49400713839122
// MI455X (gfx1250) — matrix operand order repaired and checked by reading, not compiled, not run on hardware
//
#include <hip/hip_runtime.h>

// ---------------------------------------------------------------------------
// AdvancedGNNOptimizer forward for MI455X (gfx1250, wave32, WMMA)
//
// Key optimization: per-layer edge GEMM ea_full@We (22 GF/layer) is consumed
// only through the projection onto att_edge -> precompute V = We@att_edge
// (256x8 per layer) and do ONE GEMM a_e_all = ea_full @ [V_0..V_7] (256x64).
// All GEMMs use v_wmma_f32_16x16x32_bf16 (bf16 in, f32 acc), register-tiled
// 4 row-tiles per wave so each B fragment feeds 4 WMMAs.
// ---------------------------------------------------------------------------

typedef unsigned short u16;
typedef unsigned int   u32;
typedef __attribute__((ext_vector_type(16))) __bf16 v16bf;
typedef __attribute__((ext_vector_type(8)))  float  v8f;

#define N_NODES 10000
#define N_EDGES 160000
#define N_BATCH 16
#define HDIM    256
#define HEADS   8
#define CHH     32
#define LAYERS  8
#define EALL    (N_EDGES + N_NODES)   // attention edges incl. self loops
#define MROWS_AE (N_EDGES + 16)       // ea_full rows: E real + 16x replicated mean row

__device__ __forceinline__ u16 f2bf(float f){
  u32 u = __float_as_uint(f); u += 0x7FFFu + ((u >> 16) & 1u); return (u16)(u >> 16);
}
__device__ __forceinline__ float bf2f(u16 h){ return __uint_as_float(((u32)h) << 16); }
// monotonic float<->uint mapping for atomicMax on floats (handles negatives)
__device__ __forceinline__ u32 mapf(float f){
  u32 u = __float_as_uint(f); return (u & 0x80000000u) ? ~u : (u | 0x80000000u);
}
__device__ __forceinline__ float unmapf(u32 u){
  return __uint_as_float((u & 0x80000000u) ? (u & 0x7FFFFFFFu) : ~u);
}
#define MAPF_NEG_INF 0x007FFFFFu
__device__ __forceinline__ float gelu_f(float x){
  return 0.5f * x * (1.0f + erff(x * 0.70710678118f));
}
__device__ __forceinline__ float wave_sum(float v){
  for (int m = 16; m > 0; m >>= 1) v += __shfl_xor(v, m);
  return v;
}

// ---------------------------------------------------------------------------
// Pack f32 weight B[K,Nout] into wave32 WMMA B fragments, this code's contiguous-k order (same as the A loads):
// per (nTile,kTile) a 512-u16 block; lane l holds col nTile*16+(l&15),
// K range k0+[0..15] (lanes 0-15) or k0+[16..31] (lanes 16-31), 16 bf16/lane.
// ---------------------------------------------------------------------------
__global__ void k_pack_b(const float* __restrict__ B, u16* __restrict__ P,
                         int K, int Nout){
  int gid = blockIdx.x * blockDim.x + threadIdx.x;
  if (gid >= K * Nout) return;
  int j    = gid & 15;
  int lane = (gid >> 4) & 31;
  int kt   = (gid >> 9) % (K >> 5);
  int nt   = gid / ((K >> 5) << 9);
  int col  = (nt << 4) + (lane & 15);
  int k    = (kt << 5) + ((lane >> 4) << 4) + j;
  P[gid] = f2bf(B[(size_t)k * Nout + col]);
}

// ---------------------------------------------------------------------------
// Tiled WMMA GEMM: C[M,Nout] = A[M,K]_bf16 @ Bpack + bias.
// Block = 8 waves; block tile = 64 rows x 128 cols. Each wave owns a 16-col
// strip and 4 row-tiles (4 f32 accumulators); per k-step one B fragment is
// loaded from the packed stream and reused by 4 WMMAs against 4 A fragments
// read from the 64-row LDS A panel (2x ds_load_b128 each).
// Optional epilogue: f32 store, bf16 store, column-sum atomics (for mean).
// M multiple of 16; K multiple of 32, K <= 256; Nout multiple of 16.
// ---------------------------------------------------------------------------
__global__ void __launch_bounds__(256)
k_gemm_wmma(const u16* __restrict__ A, const u16* __restrict__ P,
            float* __restrict__ Cf, u16* __restrict__ Cb,
            const float* __restrict__ bias, float* __restrict__ colsum,
            int M, int K, int Nout){
  __shared__ u16 Ap[64 * 256];
  const int tid  = threadIdx.x;
  const int row0 = blockIdx.x << 6;
  const int vr   = min(64, M - row0);          // valid rows in this panel
  { // contiguous A panel -> LDS (rows are contiguous in row-major A)
    const uint4* s = (const uint4*)(A + (size_t)row0 * K);
    uint4* d = (uint4*)Ap;
    const int n16 = (vr * K) >> 3;
    for (int i = tid; i < n16; i += 256) d[i] = s[i];
  }
  __syncthreads();
  const int wave = tid >> 5, lane = tid & 31;
  const int colTile = (blockIdx.y << 7) + (wave << 4);
  if (colTile >= Nout) return;

  const int r    = lane & 15;
  const int koff = (lane < 16) ? 0 : 16;       // A frag K split: contiguous 16 per lane half
  v8f acc[4] = {};
  const u16* pb = P + (((size_t)(colTile >> 4) * (K >> 5)) << 9) + (lane << 4);
  for (int k0 = 0; k0 < K; k0 += 32){
    union { v16bf v; uint4 q[2]; } bfr;
    bfr.q[0] = ((const uint4*)pb)[0];
    bfr.q[1] = ((const uint4*)pb)[1];
    pb += 512;
#pragma unroll
    for (int t = 0; t < 4; ++t){
      union { v16bf v; uint4 q[2]; } af;
      const u16* ap = Ap + ((t << 4) + r) * K + k0 + koff;
      af.q[0] = *(const uint4*)ap;             // K = k0+koff  .. +7
      af.q[1] = *(const uint4*)(ap + 8);       // K = k0+koff+8 .. +15
      acc[t] = __builtin_amdgcn_wmma_f32_16x16x32_bf16(
          false, af.v, false, bfr.v, (short)0, acc[t], false, false);
    }
  }
  const int mo  = (lane >= 16) ? 8 : 0;
  const int col = colTile + (lane & 15);
  const float bv = bias ? bias[col] : 0.0f;
#pragma unroll
  for (int t = 0; t < 4; ++t){
    for (int v = 0; v < 8; ++v){
      int rr = row0 + (t << 4) + v + mo;
      if (rr >= M) continue;
      float val = acc[t][v] + bv;
      size_t idx = (size_t)rr * Nout + col;
      if (Cf) Cf[idx] = val;
      if (Cb) Cb[idx] = f2bf(val);
      if (colsum) atomicAdd(&colsum[col], val);
    }
  }
}

// ---------------------------------------------------------------------------
// embed stage 1: h1 = gelu(LN(x @ W1 + b1)) -> bf16 [rows,256]; wave/row
// ---------------------------------------------------------------------------
__global__ void __launch_bounds__(256)
k_embed_l1(const float* __restrict__ X, int Fin,
           const float* __restrict__ W, const float* __restrict__ b,
           const float* __restrict__ lng, const float* __restrict__ lnb,
           u16* __restrict__ out, int rows){
  int row = (blockIdx.x << 3) + (threadIdx.x >> 5);
  if (row >= rows) return;
  int lane = threadIdx.x & 31;
  float xin[10];
  for (int k = 0; k < Fin; ++k) xin[k] = X[(size_t)row * Fin + k];
  float t[8], s = 0.f, s2 = 0.f;
  for (int i = 0; i < 8; ++i){
    int j = lane * 8 + i;
    float a = b[j];
    for (int k = 0; k < Fin; ++k) a += xin[k] * W[k * HDIM + j];
    t[i] = a; s += a; s2 += a * a;
  }
  s = wave_sum(s); s2 = wave_sum(s2);
  float m  = s * (1.0f / HDIM);
  float rs = rsqrtf(s2 * (1.0f / HDIM) - m * m + 1e-5f);
  for (int i = 0; i < 8; ++i){
    int j = lane * 8 + i;
    float y = (t[i] - m) * rs * lng[j] + lnb[j];
    out[(size_t)row * HDIM + j] = f2bf(gelu_f(y));
  }
}

__global__ void k_zero(float* p, int n){
  int g = blockIdx.x * blockDim.x + threadIdx.x; if (g < n) p[g] = 0.f;
}

// write 16 replicated mean rows (self-loop edge feature) at end of ea_full
__global__ void k_mean_rows(const float* __restrict__ colsum, u16* __restrict__ eaf){
  int g = blockIdx.x * blockDim.x + threadIdx.x;
  if (g >= 16 * HDIM) return;
  eaf[(size_t)(N_EDGES + (g >> 8)) * HDIM + (g & 255)] =
      f2bf(colsum[g & 255] * (1.0f / N_EDGES));
}

// V_l[k,hd] = sum_c We_l[k, hd*32+c] * att_edge_l[hd,c]  -> Vall[256,64]
__global__ void k_vall(const float* __restrict__ We, const float* __restrict__ attE,
                       float* __restrict__ Vall, int l){
  int g = blockIdx.x * blockDim.x + threadIdx.x;
  if (g >= HDIM * HEADS) return;
  int k = g >> 3, hd = g & 7;
  float s = 0.f;
  for (int c = 0; c < CHH; ++c) s += We[(size_t)k * HDIM + hd * CHH + c] * attE[hd * CHH + c];
  Vall[(size_t)k * 64 + l * 8 + hd] = s;
}

// a_src/a_dst[n,h] = sum_c h[n,h*32+c]*att_{src,dst}[h,c]
__global__ void k_asrcdst(const float* __restrict__ h, const float* __restrict__ attS,
                          const float* __restrict__ attD,
                          float* __restrict__ aS, float* __restrict__ aD){
  int g = blockIdx.x * blockDim.x + threadIdx.x;
  if (g >= N_NODES * HEADS) return;
  int n = g >> 3, hd = g & 7;
  const float* hp = h + (size_t)n * HDIM + hd * CHH;
  float s = 0.f, d = 0.f;
  for (int c = 0; c < CHH; ++c){ float v = hp[c]; s += v * attS[hd*CHH+c]; d += v * attD[hd*CHH+c]; }
  aS[g] = s; aD[g] = d;
}

__global__ void k_init_layer(float* __restrict__ xn_new, const float* __restrict__ bias,
                             u32* __restrict__ amaxU, float* __restrict__ asum){
  int g = blockIdx.x * blockDim.x + threadIdx.x;
  if (g >= N_NODES * HDIM) return;
  int ch = g & 255;
  xn_new[g] = bias[ch];
  if (ch < 8){ int n = g >> 8; amaxU[n*8+ch] = MAPF_NEG_INF; asum[n*8+ch] = 0.f; }
}

// pass A: alpha = leaky_relu(a_src[s]+a_dst[d]+a_e), segment max via atomicMax
__global__ void k_attn_max(const int* __restrict__ src, const int* __restrict__ dst,
                           const float* __restrict__ aS, const float* __restrict__ aD,
                           const float* __restrict__ aE, int l,
                           float* __restrict__ alpha, u32* __restrict__ amaxU){
  int g = blockIdx.x * blockDim.x + threadIdx.x;
  if (g >= EALL * HEADS) return;
  int e = g >> 3, hd = g & 7;
  int s  = (e < N_EDGES) ? src[e] : (e - N_EDGES);
  int d  = (e < N_EDGES) ? dst[e] : (e - N_EDGES);
  int ar = (e < N_EDGES) ? e : N_EDGES;       // self-loops use mean row
  float a = aS[s*8+hd] + aD[d*8+hd] + aE[(size_t)ar * 64 + l * 8 + hd];
  a = (a > 0.f) ? a : 0.2f * a;
  alpha[g] = a;
  atomicMax(&amaxU[d*8+hd], mapf(a));
}

// pass B: ex = exp(alpha - max), segment sum
__global__ void k_attn_exp(const int* __restrict__ dst, float* __restrict__ alpha,
                           const u32* __restrict__ amaxU, float* __restrict__ asum){
  int g = blockIdx.x * blockDim.x + threadIdx.x;
  if (g >= EALL * HEADS) return;
  int e = g >> 3, hd = g & 7;
  int d = (e < N_EDGES) ? dst[e] : (e - N_EDGES);
  float ex = expf(alpha[g] - unmapf(amaxU[d*8+hd]));
  alpha[g] = ex;
  atomicAdd(&asum[d*8+hd], ex);
}

// message: xn_new[d] += h[s] * att  (wave per edge, 8 channels per lane)
__global__ void __launch_bounds__(256)
k_message(const int* __restrict__ src, const int* __restrict__ dst,
          const float* __restrict__ h, const float* __restrict__ ex,
          const float* __restrict__ asum, float* __restrict__ xn_new){
  int e = (blockIdx.x << 3) + (threadIdx.x >> 5);
  if (e >= EALL) return;
  int lane = threadIdx.x & 31;
  int s = (e < N_EDGES) ? src[e] : (e - N_EDGES);
  int d = (e < N_EDGES) ? dst[e] : (e - N_EDGES);
  int hd = lane >> 2;   // 8 consecutive channels stay inside one head
  float w = ex[(size_t)e * 8 + hd] / (asum[d*8+hd] + 1e-16f);
  const float4* hp = (const float4*)(h + (size_t)s * HDIM + lane * 8);
  float4 h0 = hp[0], h1 = hp[1];
  float* op = xn_new + (size_t)d * HDIM + lane * 8;
  atomicAdd(op+0, h0.x*w); atomicAdd(op+1, h0.y*w);
  atomicAdd(op+2, h0.z*w); atomicAdd(op+3, h0.w*w);
  atomicAdd(op+4, h1.x*w); atomicAdd(op+5, h1.y*w);
  atomicAdd(op+6, h1.z*w); atomicAdd(op+7, h1.w*w);
}

// post: xn = gelu(LN(xn_new)) [+ xn_old], write f32 + bf16 (next GEMM input)
__global__ void __launch_bounds__(256)
k_post(const float* __restrict__ xn_new, const float* __restrict__ lng,
       const float* __restrict__ lnb, float* __restrict__ xn,
       u16* __restrict__ xnbf, int addRes){
  int n = (blockIdx.x << 3) + (threadIdx.x >> 5);
  if (n >= N_NODES) return;
  int lane = threadIdx.x & 31;
  float t[8], s = 0.f, s2 = 0.f;
  for (int i = 0; i < 8; ++i){
    t[i] = xn_new[(size_t)n * HDIM + lane * 8 + i];
    s += t[i]; s2 += t[i] * t[i];
  }
  s = wave_sum(s); s2 = wave_sum(s2);
  float m  = s * (1.0f / HDIM);
  float rs = rsqrtf(s2 * (1.0f / HDIM) - m * m + 1e-5f);
  for (int i = 0; i < 8; ++i){
    size_t idx = (size_t)n * HDIM + lane * 8 + i;
    float y = gelu_f((t[i] - m) * rs * lng[lane*8+i] + lnb[lane*8+i]);
    if (addRes) y += xn[idx];
    xn[idx] = y; xnbf[idx] = f2bf(y);
  }
}

__global__ void k_pool_init(float* addP, u32* maxU, float* cnt){
  int g = blockIdx.x * blockDim.x + threadIdx.x;
  if (g >= N_BATCH * HDIM) return;
  addP[g] = 0.f; maxU[g] = MAPF_NEG_INF;
  if ((g & 255) == 0) cnt[g >> 8] = 0.f;
}

__global__ void __launch_bounds__(256)
k_pool(const float* __restrict__ xn, const int* __restrict__ batch,
       float* __restrict__ addP, u32* __restrict__ maxU, float* __restrict__ cnt){
  int n = (blockIdx.x << 3) + (threadIdx.x >> 5);
  if (n >= N_NODES) return;
  int lane = threadIdx.x & 31;
  int b = batch[n];
  for (int i = 0; i < 8; ++i){
    int ch = lane * 8 + i;
    float v = xn[(size_t)n * HDIM + ch];
    atomicAdd(&addP[b * HDIM + ch], v);
    atomicMax(&maxU[b * HDIM + ch], mapf(v));
  }
  if (lane == 0) atomicAdd(&cnt[b], 1.0f);
}

__global__ void k_xg(const float* __restrict__ addP, const u32* __restrict__ maxU,
                     const float* __restrict__ cnt, float* __restrict__ xg){
  int g = blockIdx.x * blockDim.x + threadIdx.x;
  if (g >= N_BATCH * 3 * HDIM) return;
  int b = g / (3 * HDIM), j = g % (3 * HDIM);
  float v;
  if (j < HDIM)          v = addP[b * HDIM + j] / fmaxf(cnt[b], 1.0f);
  else if (j < 2 * HDIM) v = addP[b * HDIM + j - HDIM];
  else                   v = unmapf(maxU[b * HDIM + j - 2 * HDIM]);
  xg[g] = v;
}

// tiny head MLP layer: block per row; mode 0 = LN+GELU, 1 = sigmoid (final)
__global__ void __launch_bounds__(512)
k_head(const float* __restrict__ in, int Fin, int Fo,
       const float* __restrict__ W, const float* __restrict__ b,
       const float* __restrict__ g, const float* __restrict__ lb,
       float* __restrict__ out, int mode){
  __shared__ float red[512];
  int row = blockIdx.x, j = threadIdx.x;
  float a = 0.f;
  if (j < Fo){
    a = b[j];
    const float* ip = in + (size_t)row * Fin;
    for (int k = 0; k < Fin; ++k) a += ip[k] * W[(size_t)k * Fo + j];
  }
  if (mode == 1){
    if (j < Fo) out[(size_t)row * Fo + j] = 1.0f / (1.0f + expf(-a));
    return;
  }
  red[j] = (j < Fo) ? a : 0.f; __syncthreads();
  for (int st = 256; st > 0; st >>= 1){ if (j < st) red[j] += red[j + st]; __syncthreads(); }
  float mean = red[0] / Fo; __syncthreads();
  red[j] = (j < Fo) ? (a - mean) * (a - mean) : 0.f; __syncthreads();
  for (int st = 256; st > 0; st >>= 1){ if (j < st) red[j] += red[j + st]; __syncthreads(); }
  float var = red[0] / Fo;
  if (j < Fo){
    float y = (a - mean) * rsqrtf(var + 1e-5f) * g[j] + lb[j];
    out[(size_t)row * Fo + j] = gelu_f(y);
  }
}

// ---------------------------------------------------------------------------
extern "C" void kernel_launch(void* const* d_in, const int* in_sizes, int n_in,
                              void* d_out, int out_size, void* d_ws, size_t ws_size,
                              hipStream_t stream) {
  // ---- input pytree (jax dict flatten = sorted keys) ----
  const float* X     = (const float*)d_in[0];
  const int*   EI    = (const int*)  d_in[1];
  const float* EATTR = (const float*)d_in[2];
  const int*   BATCH = (const int*)  d_in[3];
  // edge_embed: 4..9 (l1.W,l1.b,l2.W,l2.b,ln.b,ln.g)
  // layers: 10 + 8*i (W,We,att_dst,att_edge,att_src,bias,ln.b,ln.g)
  // metrics_mlp: 74..87  node_embed: 88..93  param_mlp: 94..107
  const int* SRC = EI, *DST = EI + N_EDGES;

  // ---- workspace carve ----
  char* ws = (char*)d_ws;
  size_t off = 0;
  auto alloc = [&](size_t bytes)->char*{
    char* p = ws + off; off = (off + bytes + 255) & ~(size_t)255; return p;
  };
  u16*   wp     = (u16*)  alloc((size_t)(10 * 65536 + 16384) * 2); // packed weights
  float* vallf  = (float*)alloc((size_t)HDIM * 64 * 4);
  float* colsum = (float*)alloc(HDIM * 4);
  u16*   eaf    = (u16*)  alloc((size_t)MROWS_AE * HDIM * 2);      // ea_full bf16
  float* ae     = (float*)alloc((size_t)MROWS_AE * 64 * 4);        // a_e (all layers)
  float* xn     = (float*)alloc((size_t)N_NODES * HDIM * 4);
  u16*   xnbf   = (u16*)  alloc((size_t)N_NODES * HDIM * 2);
  float* hbuf   = (float*)alloc((size_t)N_NODES * HDIM * 4);
  char*  scr    = alloc((size_t)N_EDGES * HDIM * 2);               // time-shared region
  // phase 1/2: he1; phase 3: layer scratch; phase 4/5: pooling + heads
  u16*   he1    = (u16*)scr;
  float* aS     = (float*)(scr);
  float* aD     = (float*)(scr +  512 * 1024);
  u32*   amaxU  = (u32*)  (scr + 1024 * 1024);
  float* asum   = (float*)(scr + 1536 * 1024);
  float* alpha  = (float*)(scr + 2048 * 1024);
  float* xn_new = (float*)(scr + (size_t)8  * 1024 * 1024);
  float* addP   = (float*)(scr + (size_t)20 * 1024 * 1024);
  u32*   maxU   = (u32*)  (scr + (size_t)20 * 1024 * 1024 +  64 * 1024);
  float* cnt    = (float*)(scr + (size_t)20 * 1024 * 1024 + 128 * 1024);
  float* xg     = (float*)(scr + (size_t)20 * 1024 * 1024 + 192 * 1024);
  float* tmpA   = (float*)(scr + (size_t)20 * 1024 * 1024 + 256 * 1024);
  float* tmpB   = (float*)(scr + (size_t)20 * 1024 * 1024 + 320 * 1024);

  // ---- phase 0: pack weights to WMMA B-fragment layout ----
  k_pack_b<<<(65536 + 255) / 256, 256, 0, stream>>>((const float*)d_in[6],  wp + 0 * 65536, 256, 256); // edge l2
  k_pack_b<<<(65536 + 255) / 256, 256, 0, stream>>>((const float*)d_in[90], wp + 1 * 65536, 256, 256); // node l2
  for (int i = 0; i < LAYERS; ++i)
    k_pack_b<<<(65536 + 255) / 256, 256, 0, stream>>>((const float*)d_in[10 + 8 * i],
                                                      wp + (2 + i) * 65536, 256, 256);
  for (int i = 0; i < LAYERS; ++i)   // Vall = [We_i @ att_edge_i]
    k_vall<<<(HDIM * HEADS + 255) / 256, 256, 0, stream>>>(
        (const float*)d_in[10 + 8 * i + 1], (const float*)d_in[10 + 8 * i + 3], vallf, i);
  k_pack_b<<<(16384 + 255) / 256, 256, 0, stream>>>(vallf, wp + 10 * 65536, 256, 64);

  // ---- phase 1: edge embed + attention-edge projection ----
  k_embed_l1<<<(N_EDGES + 7) / 8, 256, 0, stream>>>(
      EATTR, 4, (const float*)d_in[4], (const float*)d_in[5],
      (const float*)d_in[9], (const float*)d_in[8], he1, N_EDGES);
  k_zero<<<1, 256, 0, stream>>>(colsum, HDIM);
  k_gemm_wmma<<<dim3((N_EDGES + 63) / 64, 2), 256, 0, stream>>>(
      he1, wp + 0 * 65536, nullptr, eaf, (const float*)d_in[7], colsum,
      N_EDGES, 256, 256);
  k_mean_rows<<<16, 256, 0, stream>>>(colsum, eaf);
  k_gemm_wmma<<<dim3((MROWS_AE + 63) / 64, 1), 256, 0, stream>>>(
      eaf, wp + 10 * 65536, ae, nullptr, nullptr, nullptr,
      MROWS_AE, 256, 64);

  // ---- phase 2: node embed ----
  k_embed_l1<<<(N_NODES + 7) / 8, 256, 0, stream>>>(
      X, 10, (const float*)d_in[88], (const float*)d_in[89],
      (const float*)d_in[93], (const float*)d_in[92], he1, N_NODES);
  k_gemm_wmma<<<dim3((N_NODES + 63) / 64, 2), 256, 0, stream>>>(
      he1, wp + 1 * 65536, xn, xnbf, (const float*)d_in[91], nullptr,
      N_NODES, 256, 256);

  // ---- phase 3: 8 GAT layers ----
  for (int i = 0; i < LAYERS; ++i){
    const float* attS = (const float*)d_in[10 + 8 * i + 4];
    const float* attD = (const float*)d_in[10 + 8 * i + 2];
    const float* bia  = (const float*)d_in[10 + 8 * i + 5];
    const float* lnb  = (const float*)d_in[10 + 8 * i + 6];
    const float* lng  = (const float*)d_in[10 + 8 * i + 7];
    k_gemm_wmma<<<dim3((N_NODES + 63) / 64, 2), 256, 0, stream>>>(
        xnbf, wp + (2 + i) * 65536, hbuf, nullptr, nullptr, nullptr,
        N_NODES, 256, 256);
    k_asrcdst<<<(N_NODES * HEADS + 255) / 256, 256, 0, stream>>>(hbuf, attS, attD, aS, aD);
    k_init_layer<<<(N_NODES * HDIM) / 256, 256, 0, stream>>>(xn_new, bia, amaxU, asum);
    k_attn_max<<<(EALL * HEADS + 255) / 256, 256, 0, stream>>>(SRC, DST, aS, aD, ae, i, alpha, amaxU);
    k_attn_exp<<<(EALL * HEADS + 255) / 256, 256, 0, stream>>>(DST, alpha, amaxU, asum);
    k_message<<<(EALL + 7) / 8, 256, 0, stream>>>(SRC, DST, hbuf, alpha, asum, xn_new);
    k_post<<<(N_NODES + 7) / 8, 256, 0, stream>>>(xn_new, lng, lnb, xn, xnbf, i > 0 ? 1 : 0);
  }

  // ---- phase 4: global pooling ----
  k_pool_init<<<16, 256, 0, stream>>>(addP, maxU, cnt);
  k_pool<<<(N_NODES + 7) / 8, 256, 0, stream>>>(xn, BATCH, addP, maxU, cnt);
  k_xg<<<(N_BATCH * 3 * HDIM + 255) / 256, 256, 0, stream>>>(addP, maxU, cnt, xg);

  // ---- phase 5: two heads (param_mlp @94 first, metrics_mlp @74) ----
  for (int p = 0; p < 2; ++p){
    int base = (p == 0) ? 94 : 74;
    float* outp = (float*)d_out + (p == 0 ? 0 : N_BATCH * 3);
    k_head<<<N_BATCH, 512, 0, stream>>>(xg, 768, 512,
        (const float*)d_in[base + 0], (const float*)d_in[base + 1],
        (const float*)d_in[base + 9], (const float*)d_in[base + 8], tmpA, 0);
    k_head<<<N_BATCH, 512, 0, stream>>>(tmpA, 512, 256,
        (const float*)d_in[base + 2], (const float*)d_in[base + 3],
        (const float*)d_in[base + 11], (const float*)d_in[base + 10], tmpB, 0);
    k_head<<<N_BATCH, 512, 0, stream>>>(tmpB, 256, 128,
        (const float*)d_in[base + 4], (const float*)d_in[base + 5],
        (const float*)d_in[base + 13], (const float*)d_in[base + 12], tmpA, 0);
    k_head<<<N_BATCH, 512, 0, stream>>>(tmpA, 128, 3,
        (const float*)d_in[base + 6], (const float*)d_in[base + 7],
        nullptr, nullptr, outp, 1);
  }
  (void)in_sizes; (void)n_in; (void)out_size; (void)ws_size;
}
